// SASAStem2d_7198365188212
// MI455X (gfx1250) — matrix operand order repaired and checked by reading, not compiled, not run on hardware
//
#include <hip/hip_runtime.h>
#include <hip/hip_bf16.h>

// ---------------------------------------------------------------------------
// SASA stem, MI455X (gfx1250, wave32).
//  Stage 1: prep     — convert x -> bf16 pixel-major, concat W -> bf16,
//                      mixture softmax w_mix[4][49].
//  Stage 2: qkv_gemm — fused [768 x 64] x [64 x 6272] GEMM via
//                      v_wmma_f32_16x16x32_bf16 (2 WMMA per 16x16 tile).
//                      Outputs stored pixel-major for vectorized attention.
//  Stage 3: attn     — per-(n,head,pixel) 7x7 local attention on VALU,
//                      scores kept in registers (49-wide unroll), fast exp.
// ---------------------------------------------------------------------------

#define N_IMG   2
#define C_IN    64
#define HS      56
#define WS_DIM  56
#define HW      (HS * WS_DIM)          // 3136
#define C_OUT   128
#define HEADS   8
#define DHEAD   (C_OUT / HEADS)        // 16
#define KWIN    7
#define UWIN    (KWIN * KWIN)          // 49
#define MIX     4
#define ROWS_ALL (C_OUT + C_OUT + MIX * C_OUT)  // 768 (q | k | v)

typedef __attribute__((ext_vector_type(16))) __bf16 v16bf;
typedef __attribute__((ext_vector_type(8)))  __bf16 v8bf;
typedef __attribute__((ext_vector_type(8)))  float  v8f;
typedef __attribute__((ext_vector_type(4)))  float  vf4;

// Workspace layout (bytes); all offsets 256B aligned.
#define OFF_XT   (0)                                        // bf16 [N][HW][64]
#define SZ_XT    ((size_t)N_IMG * HW * C_IN * 2)            // 802,816
#define OFF_W    (OFF_XT + SZ_XT)                           // bf16 [768][64]
#define SZ_W     ((size_t)ROWS_ALL * C_IN * 2)              //    98,304
#define OFF_WMIX (OFF_W + SZ_W)                             // f32 [4][49]
#define OFF_QT   (OFF_WMIX + 1024)                          // f32 [N][HW][128]
#define SZ_QK    ((size_t)N_IMG * HW * C_OUT * 4)           // 3,211,264
#define OFF_KT   (OFF_QT + SZ_QK)
#define OFF_VT   (OFF_KT + SZ_QK)                           // f32 [N][HW][512]

// ---------------------------------------------------------------------------
// Stage 1: conversions + mixture softmax.
// ---------------------------------------------------------------------------
__global__ __launch_bounds__(256) void sasa_prep(
    const float* __restrict__ x,      // [N][64][HW]
    const float* __restrict__ Wq,     // [128][64]
    const float* __restrict__ Wk,     // [128][64]
    const float* __restrict__ Wv,     // [512][64]
    const float* __restrict__ row_emb,// [128][7]
    const float* __restrict__ col_emb,// [128][7]
    const float* __restrict__ mix_emb,// [128][4]
    __bf16* __restrict__ xT,          // [N][HW][64]
    __bf16* __restrict__ wcat,        // [768][64]
    float*  __restrict__ wmix)        // [4][49]
{
  const int gid = blockIdx.x * blockDim.x + threadIdx.x;

  // x [n][c][f] -> xT [n][f][c] (bf16). Coalesced read, 128B-strided write.
  if (gid < N_IMG * C_IN * HW) {
    const int n = gid / (C_IN * HW);
    const int r = gid - n * (C_IN * HW);
    const int c = r / HW;
    const int f = r - c * HW;
    xT[((size_t)n * HW + f) * C_IN + c] = (__bf16)x[gid];
  }

  // Concatenated weights: rows [0,128)=Wq, [128,256)=Wk, [256,768)=Wv.
  if (gid < ROWS_ALL * C_IN) {
    const int row = gid / C_IN;
    const int c   = gid - row * C_IN;
    float v;
    if (row < C_OUT)          v = Wq[row * C_IN + c];
    else if (row < 2 * C_OUT) v = Wk[(row - C_OUT) * C_IN + c];
    else                      v = Wv[(row - 2 * C_OUT) * C_IN + c];
    wcat[gid] = (__bf16)v;
  }

  // w_mix[m][u] = softmax_m( (mix^T row)[m,ky] + (mix^T col)[m,kx] ).
  if (gid < UWIN) {
    const int ky = gid / KWIN, kx = gid - ky * KWIN;
    float s[MIX];
#pragma unroll
    for (int m = 0; m < MIX; ++m) {
      float rr = 0.f, cc = 0.f;
      for (int c = 0; c < C_OUT; ++c) {
        const float me = mix_emb[c * MIX + m];
        rr = fmaf(me, row_emb[c * KWIN + ky], rr);
        cc = fmaf(me, col_emb[c * KWIN + kx], cc);
      }
      s[m] = rr + cc;
    }
    float mx = fmaxf(fmaxf(s[0], s[1]), fmaxf(s[2], s[3]));
    float e[MIX], den = 0.f;
#pragma unroll
    for (int m = 0; m < MIX; ++m) { e[m] = __expf(s[m] - mx); den += e[m]; }
    const float inv = 1.f / den;
#pragma unroll
    for (int m = 0; m < MIX; ++m) wmix[m * UWIN + gid] = e[m] * inv;
  }
}

// ---------------------------------------------------------------------------
// Stage 2: fused QKV projection GEMM (WMMA bf16 -> f32).
// grid = (HW/16 = 196, 48/8 = 6, N_IMG); 256 threads = 8 waves;
// each wave owns one 16(oc) x 16(pixel) tile, K=64 -> 2 WMMA ops.
// ---------------------------------------------------------------------------
__global__ __launch_bounds__(256) void sasa_qkv_gemm(
    const __bf16* __restrict__ xT,    // [N][HW][64]
    const __bf16* __restrict__ wcat,  // [768][64]
    float* __restrict__ qT,           // [N][HW][128]
    float* __restrict__ kT,           // [N][HW][128]
    float* __restrict__ vT)           // [N][HW][512]
{
  const int lane    = threadIdx.x & 31;
  const int wave    = threadIdx.x >> 5;
  const int colTile = blockIdx.x;                 // pixel tile, 0..195
  const int rowTile = blockIdx.y * 8 + wave;      // oc tile,    0..47
  const int nimg    = blockIdx.z;

  const int l15 = lane & 15;   // A: row M; B/C/D: column N (pixel)
  const int kh  = lane >> 4;   // K-half selector per ISA layouts

  const int pix = colTile * 16 + l15;
  const __bf16* wrow = wcat + (size_t)(rowTile * 16 + l15) * C_IN;
  const __bf16* xrow = xT + ((size_t)nimg * HW + pix) * C_IN;

  v8f acc = {};
#pragma unroll
  for (int k0 = 0; k0 < C_IN; k0 += 32) {
    // A (16x32 bf16): lane group kh holds K = k0 + kh*16 + e, contiguous in wcat.
    const v16bf a = *(const v16bf*)(wrow + k0 + kh * 16);
    // B (32x16 bf16): lane group kh holds K = k0 + kh*16 + e, contiguous in xT.
    const v16bf b = *(const v16bf*)(xrow + k0 + kh * 16);
    acc = __builtin_amdgcn_wmma_f32_16x16x32_bf16(
        /*neg_a=*/false, a, /*neg_b=*/false, b,
        /*c_mod=*/(short)0, acc, /*reuse_a=*/false, /*reuse_b=*/false);
  }

  // D layout: VGPR r -> M = r + 8*kh; so this lane writes 8 consecutive
  // output channels starting at oc0. Route to q/k/v, store pixel-major.
  int oc0 = rowTile * 16 + kh * 8;
  float* dst; int ch;
  if (oc0 < C_OUT)          { dst = qT;              ch = C_OUT; }
  else if (oc0 < 2 * C_OUT) { dst = kT; oc0 -= C_OUT;     ch = C_OUT; }
  else                      { dst = vT; oc0 -= 2 * C_OUT; ch = MIX * C_OUT; }
  float* p = dst + ((size_t)nimg * HW + pix) * ch + oc0;
  vf4 lo = {acc[0], acc[1], acc[2], acc[3]};
  vf4 hi = {acc[4], acc[5], acc[6], acc[7]};
  *(vf4*)(p)     = lo;
  *(vf4*)(p + 4) = hi;
}

// ---------------------------------------------------------------------------
// Stage 3: fused 7x7 local attention. One thread per (n, head, pixel).
// Scores fully register-resident (unrolled u-loop), zero-padding semantics:
// out-of-bounds neighbors contribute score 0 to the softmax and v = 0.
// ---------------------------------------------------------------------------
__global__ __launch_bounds__(256) void sasa_attn(
    const float* __restrict__ qT,    // [N][HW][128]
    const float* __restrict__ kT,    // [N][HW][128]
    const float* __restrict__ vT,    // [N][HW][512]
    const float* __restrict__ wmix,  // [4][49]
    float* __restrict__ out)         // [N][128][HW]
{
  const int gid  = blockIdx.x * blockDim.x + threadIdx.x;   // 0 .. N*HEADS*HW
  const int pix  = gid % HW;
  const int nh   = gid / HW;
  const int h    = nh % HEADS;
  const int nimg = nh / HEADS;
  const int y = pix / WS_DIM, xq = pix - y * WS_DIM;

  const float* qv = qT + ((size_t)nimg * HW + pix) * C_OUT + h * DHEAD;
  float q[DHEAD];
#pragma unroll
  for (int d = 0; d < DHEAD; d += 4) {
    vf4 t = *(const vf4*)(qv + d);
    q[d] = t[0]; q[d+1] = t[1]; q[d+2] = t[2]; q[d+3] = t[3];
  }

  float s[UWIN];
  float smax = 0.f;   // zero-padded neighbors always yield score 0
#pragma unroll
  for (int u = 0; u < UWIN; ++u) {
    const int dy = u / KWIN - 3, dx = u % KWIN - 3;
    const int yy = y + dy, xx = xq + dx;
    float sc = 0.f;
    if ((unsigned)yy < (unsigned)HS && (unsigned)xx < (unsigned)WS_DIM) {
      const float* kv = kT + ((size_t)nimg * HW + yy * WS_DIM + xx) * C_OUT + h * DHEAD;
#pragma unroll
      for (int d = 0; d < DHEAD; d += 4) {
        vf4 t = *(const vf4*)(kv + d);
        sc = fmaf(q[d], t[0], sc); sc = fmaf(q[d+1], t[1], sc);
        sc = fmaf(q[d+2], t[2], sc); sc = fmaf(q[d+3], t[3], sc);
      }
    }
    s[u] = sc;
    smax = fmaxf(smax, sc);
  }
  float den = 0.f;
#pragma unroll
  for (int u = 0; u < UWIN; ++u) { s[u] = __expf(s[u] - smax); den += s[u]; }
  const float inv = 1.f / den;

  float acc[DHEAD];
#pragma unroll
  for (int d = 0; d < DHEAD; ++d) acc[d] = 0.f;

#pragma unroll
  for (int u = 0; u < UWIN; ++u) {
    const int dy = u / KWIN - 3, dx = u % KWIN - 3;
    const int yy = y + dy, xx = xq + dx;
    if ((unsigned)yy < (unsigned)HS && (unsigned)xx < (unsigned)WS_DIM) {
      const float a = s[u] * inv;
      const float* vbase =
          vT + ((size_t)nimg * HW + yy * WS_DIM + xx) * (MIX * C_OUT) + h * DHEAD;
#pragma unroll
      for (int m = 0; m < MIX; ++m) {
        const float wm = a * wmix[m * UWIN + u];
        const float* vv = vbase + m * C_OUT;
#pragma unroll
        for (int d = 0; d < DHEAD; d += 4) {
          vf4 t = *(const vf4*)(vv + d);
          acc[d]   = fmaf(wm, t[0], acc[d]);
          acc[d+1] = fmaf(wm, t[1], acc[d+1]);
          acc[d+2] = fmaf(wm, t[2], acc[d+2]);
          acc[d+3] = fmaf(wm, t[3], acc[d+3]);
        }
      }
    }
  }

  float* op = out + ((size_t)nimg * C_OUT + h * DHEAD) * HW + pix;
#pragma unroll
  for (int d = 0; d < DHEAD; ++d) op[(size_t)d * HW] = acc[d];
}

// ---------------------------------------------------------------------------
extern "C" void kernel_launch(void* const* d_in, const int* in_sizes, int n_in,
                              void* d_out, int out_size, void* d_ws, size_t ws_size,
                              hipStream_t stream) {
  const float* x       = (const float*)d_in[0];
  const float* Wq      = (const float*)d_in[1];
  const float* Wk      = (const float*)d_in[2];
  const float* Wv      = (const float*)d_in[3];
  const float* row_emb = (const float*)d_in[4];
  const float* col_emb = (const float*)d_in[5];
  const float* mix_emb = (const float*)d_in[6];
  float* out = (float*)d_out;

  char* ws = (char*)d_ws;
  __bf16* xT   = (__bf16*)(ws + OFF_XT);
  __bf16* wcat = (__bf16*)(ws + OFF_W);
  float*  wmix = (float*)(ws + OFF_WMIX);
  float*  qT   = (float*)(ws + OFF_QT);
  float*  kT   = (float*)(ws + OFF_KT);
  float*  vT   = (float*)(ws + OFF_VT);

  // Stage 1: 401408 conversion items -> 1568 blocks x 256.
  sasa_prep<<<dim3((N_IMG * C_IN * HW + 255) / 256), 256, 0, stream>>>(
      x, Wq, Wk, Wv, row_emb, col_emb, mix_emb, xT, wcat, wmix);

  // Stage 2: 196 pixel tiles x 6 row-supertiles (8 waves each) x N images.
  sasa_qkv_gemm<<<dim3(HW / 16, (ROWS_ALL / 16) / 8, N_IMG), 256, 0, stream>>>(
      xT, wcat, qT, kT, vT);

  // Stage 3: one thread per (n, head, pixel) = 50176 threads.
  sasa_attn<<<dim3((N_IMG * HEADS * HW) / 256), 256, 0, stream>>>(
      qT, kT, vT, wmix, out);
}
